// ColumnGAT_60232621359200
// MI455X (gfx1250) — matrix operand order repaired and checked by reading, not compiled, not run on hardware
//
#include <hip/hip_runtime.h>
#include <hip/hip_bf16.h>

#define NN 50000
#define EE 800000
#define GG 16

typedef __attribute__((ext_vector_type(16))) __bf16 v16bf;
typedef __attribute__((ext_vector_type(8)))  float  v8f;

union FragBF { uint4 u[2]; v16bf v; };

// ---------------------------------------------------------------- utilities
__global__ void fill_kernel(float* __restrict__ p, float v, long n) {
  long i = (long)blockIdx.x * blockDim.x + threadIdx.x;
  if (i < n) p[i] = v;
}

__global__ void bias_rows_kernel(float* __restrict__ out, const float* __restrict__ bias,
                                 int F, long n) {
  long i = (long)blockIdx.x * blockDim.x + threadIdx.x;
  if (i < n) out[i] = bias[(int)(i % F)];
}

__global__ void f32_to_bf16_kernel(const float* __restrict__ in, __bf16* __restrict__ out,
                                   long n) {
  long i = (long)blockIdx.x * blockDim.x + threadIdx.x;
  if (i < n) out[i] = (__bf16)in[i];
}

__global__ void elu_kernel(float* __restrict__ p, long n) {
  long i = (long)blockIdx.x * blockDim.x + threadIdx.x;
  if (i < n) { float v = p[i]; p[i] = (v > 0.f) ? v : (__expf(v) - 1.f); }
}

__device__ inline void atomicMaxF(float* addr, float val) {
  if (val >= 0.f) atomicMax((int*)addr, __float_as_int(val));
  else            atomicMin((unsigned int*)addr, __float_as_uint(val));
}

// --------------------------------------------------- weight pack for B-frag
// Bpack[kstep][ntile][lane][j] = bf16( W[kstep*32 + (lane>>4)*8 + (j&7) + (j>>3)*16][ntile*16 + (lane&15)] )
__global__ void pack_B_kernel(const float* __restrict__ W, __bf16* __restrict__ Bp,
                              int K, int NC) {
  int idx = blockIdx.x * blockDim.x + threadIdx.x;
  int total = K * NC;
  if (idx >= total) return;
  int j     = idx & 15;
  int lane  = (idx >> 4) & 31;
  int grp   = idx >> 9;
  int ntiles = NC >> 4;
  int nt    = grp % ntiles;
  int kstep = grp / ntiles;
  int kb = kstep * 32 + ((lane >> 4) << 3) + (j & 7) + ((j >> 3) << 4);
  int n  = (nt << 4) + (lane & 15);
  Bp[idx] = (__bf16)W[(size_t)kb * NC + n];
}

// ------------------------------------------------------- WMMA bf16 GEMM f32
// C[M,NC] = A[M,K](bf16) @ Bpack(bf16) (+ bias).  One wave -> one 16x16 tile.
__global__ void wmma_gemm_bf16(const __bf16* __restrict__ A, const __bf16* __restrict__ Bp,
                               const float* __restrict__ bias, float* __restrict__ C,
                               int M, int K, int NC) {
  const int lane = threadIdx.x & 31;
  const int wave = blockIdx.x * (blockDim.x >> 5) + (threadIdx.x >> 5);
  const int ntiles = NC >> 4;
  const int mt = wave / ntiles;
  const int nt = wave - mt * ntiles;
  if (mt >= (M >> 4)) return;   // wave-uniform: EXEC stays all-ones for WMMA

  const int row  = (mt << 4) + (lane & 15);
  const int koff = ((lane >> 4) << 3);            // 0 or 8
  const __bf16* arow = A + (size_t)row * K;
  const __bf16* bp   = Bp + (((size_t)nt * 32 + lane) << 4);
  const size_t  bstep = (size_t)ntiles << 9;      // ntiles*512 elems / k-step

  v8f acc = {};
  for (int k = 0; k < K; k += 32) {
    FragBF fa, fb;
    fa.u[0] = *(const uint4*)(arow + k + koff);
    fa.u[1] = *(const uint4*)(arow + k + koff + 16);
    fb.u[0] = ((const uint4*)bp)[0];
    fb.u[1] = ((const uint4*)bp)[1];
    bp += bstep;
    acc = __builtin_amdgcn_wmma_f32_16x16x32_bf16(false, fa.v, false, fb.v,
                                                  (short)0, acc, false, false);
  }
  const int om = (mt << 4) + ((lane >> 4) << 3);
  const int on = (nt << 4) + (lane & 15);
  const float bv = bias ? bias[on] : 0.0f;
  float* crow = C + (size_t)om * NC + on;
#pragma unroll
  for (int v = 0; v < 8; ++v) crow[(size_t)v * NC] = acc[v] + bv;
}

// ----------------------------------------------- per-node attention scores
// one wave per node; lanes split the Hh*64 channels; shfl reduce per head
__global__ void att_scores_kernel(const float* __restrict__ h, const float* __restrict__ att_s,
                                  const float* __restrict__ att_d, float* __restrict__ asb,
                                  float* __restrict__ adb, int n_nodes, int Hh) {
  const int lane = threadIdx.x & 31;
  const int node = blockIdx.x * (blockDim.x >> 5) + (threadIdx.x >> 5);
  if (node >= n_nodes) return;
  const int F = Hh << 6;
  const int per = F >> 5;           // 8 for H=4, 2 for H=1
  const int base = lane * per;
  const float* hr = h + (size_t)node * F;
  float ss = 0.f, sd = 0.f;
  for (int i = 0; i < per; ++i) {
    float v = hr[base + i];
    ss += v * att_s[base + i];
    sd += v * att_d[base + i];
  }
  const int lph = 32 / Hh;          // lanes per head
  for (int off = lph >> 1; off > 0; off >>= 1) {
    ss += __shfl_xor(ss, off, 32);
    sd += __shfl_xor(sd, off, 32);
  }
  if ((lane & (lph - 1)) == 0) {
    int hh = lane / lph;
    asb[(size_t)node * Hh + hh] = ss;
    adb[(size_t)node * Hh + hh] = sd;
  }
}

// -------------------------------------------------------- edge scalar pass
__global__ void edge_max_kernel(const int* __restrict__ ei, const float* __restrict__ asb,
                                const float* __restrict__ adb, float* __restrict__ ebuf,
                                float* __restrict__ mmax, int E_, int n_nodes, int Hh) {
  long t = (long)blockIdx.x * blockDim.x + threadIdx.x;
  long total = (long)(E_ + n_nodes) * Hh;
  if (t >= total) return;
  int hh = (int)(t % Hh);
  int e  = (int)(t / Hh);
  int src, dst;
  if (e < E_) { src = ei[e]; dst = ei[E_ + e]; } else { src = dst = e - E_; }
  float v = asb[(size_t)src * Hh + hh] + adb[(size_t)dst * Hh + hh];
  v = (v >= 0.f) ? v : 0.2f * v;    // leaky_relu
  ebuf[t] = v;
  atomicMaxF(&mmax[(size_t)dst * Hh + hh], v);
}

__global__ void edge_exp_kernel(const int* __restrict__ ei, float* __restrict__ ebuf,
                                const float* __restrict__ mmax, float* __restrict__ den,
                                int E_, int n_nodes, int Hh) {
  long t = (long)blockIdx.x * blockDim.x + threadIdx.x;
  long total = (long)(E_ + n_nodes) * Hh;
  if (t >= total) return;
  int hh = (int)(t % Hh);
  int e  = (int)(t / Hh);
  int dst = (e < E_) ? ei[E_ + e] : (e - E_);
  float ex = __expf(ebuf[t] - mmax[(size_t)dst * Hh + hh]);
  ebuf[t] = ex;
  atomicAdd(&den[(size_t)dst * Hh + hh], ex);
}

// ---------------------------------------------------- message gather+scatter
// one wave per edge; 128-bit gathers of h[src], f32 atomic scatter to out[dst]
template <int HH>
__global__ void edge_scatter_kernel(const int* __restrict__ ei, const float* __restrict__ hlin,
                                    const float* __restrict__ ebuf, const float* __restrict__ den,
                                    float* __restrict__ outb, int E_, int n_nodes) {
  const int lane = threadIdx.x & 31;
  const int e = blockIdx.x * (blockDim.x >> 5) + (threadIdx.x >> 5);
  if (e >= E_ + n_nodes) return;
  int src, dst;
  if (e < E_) { src = ei[e]; dst = ei[E_ + e]; } else { src = dst = e - E_; }
  constexpr int F = HH * 64;
  constexpr int per = F / 32;
  const int base = lane * per;
  const int hh = base >> 6;
  const float a = ebuf[(size_t)e * HH + hh] / (den[(size_t)dst * HH + hh] + 1e-16f);
  const float* hs = hlin + (size_t)src * F + base;
  float* od = outb + (size_t)dst * F + base;
  float v[per];
#pragma unroll
  for (int i = 0; i < per; ++i) v[i] = hs[i];
#pragma unroll
  for (int i = 0; i < per; ++i) atomicAdd(&od[i], v[i] * a);
}

// ------------------------------------------------------------- graph pooling
__global__ void pool_kernel(const float* __restrict__ h3, const int* __restrict__ batch,
                            float* __restrict__ sums, float* __restrict__ cnts, int n_nodes) {
  __shared__ float ls[GG * 64];
  __shared__ float lc[GG];
  for (int i = threadIdx.x; i < GG * 64; i += blockDim.x) ls[i] = 0.f;
  if (threadIdx.x < GG) lc[threadIdx.x] = 0.f;
  __syncthreads();
  const int lane = threadIdx.x & 31;
  const int wpb = blockDim.x >> 5;
  for (int i = blockIdx.x * wpb + (threadIdx.x >> 5); i < n_nodes; i += gridDim.x * wpb) {
    int g = batch[i];
    const float* hr = h3 + (size_t)i * 64;
    atomicAdd(&ls[g * 64 + lane], hr[lane]);
    atomicAdd(&ls[g * 64 + 32 + lane], hr[32 + lane]);
    if (lane == 0) atomicAdd(&lc[g], 1.0f);
  }
  __syncthreads();
  for (int i = threadIdx.x; i < GG * 64; i += blockDim.x) atomicAdd(&sums[i], ls[i]);
  if (threadIdx.x < GG) atomicAdd(&cnts[threadIdx.x], lc[threadIdx.x]);
}

// ------------------------------------------------------------------ decoder
__global__ void decoder_kernel(const float* __restrict__ sums, const float* __restrict__ cnts,
                               const float* __restrict__ gf, const float* __restrict__ Wg,
                               const float* __restrict__ bg, const float* __restrict__ Wd1,
                               const float* __restrict__ bd1, const float* __restrict__ gamma,
                               const float* __restrict__ beta, const float* __restrict__ Wd2,
                               const float* __restrict__ bd2, float* __restrict__ out) {
  __shared__ float z[GG][128];
  __shared__ float z2[GG][64];
  __shared__ float logits[GG][256];
  __shared__ float rmax[GG], rsum[GG];
  const int t = threadIdx.x;
  for (int i = t; i < GG * 64; i += 256) {
    int r = i >> 6, c = i & 63;
    z[r][c] = sums[i] / fmaxf(cnts[r], 1.0f);
    float acc = bg[c];
    for (int k = 0; k < 4; ++k) acc += gf[r * 4 + k] * Wg[k * 64 + c];
    z[r][64 + c] = fmaxf(acc, 0.0f);
  }
  __syncthreads();
  for (int i = t; i < GG * 64; i += 256) {
    int r = i >> 6, c = i & 63;
    float acc = bd1[c];
    for (int k = 0; k < 128; ++k) acc += z[r][k] * Wd1[k * 64 + c];
    acc = acc * gamma[c] + beta[c];
    z2[r][c] = fmaxf(acc, 0.0f);
  }
  __syncthreads();
  for (int i = t; i < GG * 256; i += 256) {
    int r = i >> 8, c = i & 255;
    float acc = bd2[c];
    for (int k = 0; k < 64; ++k) acc += z2[r][k] * Wd2[k * 256 + c];
    logits[r][c] = acc;
  }
  __syncthreads();
  if (t < GG) {
    float mx = -1e30f;
    for (int c = 0; c < 256; ++c) mx = fmaxf(mx, logits[t][c]);
    float s = 0.f;
    for (int c = 0; c < 256; ++c) s += __expf(logits[t][c] - mx);
    rmax[t] = mx; rsum[t] = s;
  }
  __syncthreads();
  for (int i = t; i < GG * 256; i += 256) {
    int r = i >> 8, c = i & 255;
    out[i] = __expf(logits[r][c] - rmax[r]) / rsum[r];
  }
}

static inline int cdiv(long a, long b) { return (int)((a + b - 1) / b); }

extern "C" void kernel_launch(void* const* d_in, const int* in_sizes, int n_in,
                              void* d_out, int out_size, void* d_ws, size_t ws_size,
                              hipStream_t stream) {
  (void)in_sizes; (void)n_in; (void)out_size; (void)ws_size;
  const float* x    = (const float*)d_in[0];
  const int*   ei   = (const int*)d_in[1];
  const int*   batch= (const int*)d_in[2];
  const float* gf   = (const float*)d_in[3];
  const float* Wenc = (const float*)d_in[4];
  const float* benc = (const float*)d_in[5];
  const float* W1   = (const float*)d_in[6];
  const float* as1  = (const float*)d_in[7];
  const float* ad1  = (const float*)d_in[8];
  const float* b1   = (const float*)d_in[9];
  const float* W2   = (const float*)d_in[10];
  const float* as2  = (const float*)d_in[11];
  const float* ad2  = (const float*)d_in[12];
  const float* b2   = (const float*)d_in[13];
  const float* W3   = (const float*)d_in[14];
  const float* as3  = (const float*)d_in[15];
  const float* ad3  = (const float*)d_in[16];
  const float* b3   = (const float*)d_in[17];
  const float* Wg   = (const float*)d_in[18];
  const float* bg   = (const float*)d_in[19];
  const float* Wd1  = (const float*)d_in[20];
  const float* bd1  = (const float*)d_in[21];
  const float* gamma= (const float*)d_in[22];
  const float* beta = (const float*)d_in[23];
  const float* Wd2  = (const float*)d_in[24];
  const float* bd2  = (const float*)d_in[25];
  float* out = (float*)d_out;

  char* ws = (char*)d_ws;
  auto carve = [&](size_t bytes) -> char* {
    char* p = ws; ws += (bytes + 255) & ~(size_t)255; return p;
  };
  float*  hA    = (float*) carve((size_t)NN * 256 * 4);
  float*  hB    = (float*) carve((size_t)NN * 256 * 4);
  float*  hC    = (float*) carve((size_t)NN * 256 * 4);
  __bf16* Abf   = (__bf16*)carve((size_t)NN * 256 * 2);
  __bf16* Bpack = (__bf16*)carve((size_t)256 * 256 * 2);
  float*  ebuf  = (float*) carve((size_t)(EE + NN) * 4 * 4);
  float*  asb   = (float*) carve((size_t)NN * 4 * 4);
  float*  adb   = (float*) carve((size_t)NN * 4 * 4);
  float*  mmax  = (float*) carve((size_t)NN * 4 * 4);
  float*  den   = (float*) carve((size_t)NN * 4 * 4);
  float*  sums  = (float*) carve((size_t)GG * 64 * 4);
  float*  cnts  = (float*) carve((size_t)GG * 4);

  const int B = 256;
  const int Etot = EE + NN;

  auto run_conv = [&](const float* featIn, int Fin, const float* W, const float* atts,
                      const float* attd, const float* bias, int heads,
                      float* hlin, float* outBuf) {
    long nA = (long)NN * Fin;
    f32_to_bf16_kernel<<<cdiv(nA, B), B, 0, stream>>>(featIn, Abf, nA);
    int NC = heads * 64;
    pack_B_kernel<<<cdiv((long)Fin * NC, B), B, 0, stream>>>(W, Bpack, Fin, NC);
    int tiles = (NN / 16) * (NC / 16);
    wmma_gemm_bf16<<<cdiv(tiles, 8), B, 0, stream>>>(Abf, Bpack, nullptr, hlin, NN, Fin, NC);
    att_scores_kernel<<<cdiv(NN, 8), B, 0, stream>>>(hlin, atts, attd, asb, adb, NN, heads);
    long nm = (long)NN * heads;
    fill_kernel<<<cdiv(nm, B), B, 0, stream>>>(mmax, -1e30f, nm);
    fill_kernel<<<cdiv(nm, B), B, 0, stream>>>(den, 0.0f, nm);
    long nOut = (long)NN * NC;
    bias_rows_kernel<<<cdiv(nOut, B), B, 0, stream>>>(outBuf, bias, NC, nOut);
    long nEH = (long)Etot * heads;
    edge_max_kernel<<<cdiv(nEH, B), B, 0, stream>>>(ei, asb, adb, ebuf, mmax, EE, NN, heads);
    edge_exp_kernel<<<cdiv(nEH, B), B, 0, stream>>>(ei, ebuf, mmax, den, EE, NN, heads);
    int sblk = cdiv((long)Etot, 8);
    if (heads == 4)
      edge_scatter_kernel<4><<<sblk, B, 0, stream>>>(ei, hlin, ebuf, den, outBuf, EE, NN);
    else
      edge_scatter_kernel<1><<<sblk, B, 0, stream>>>(ei, hlin, ebuf, den, outBuf, EE, NN);
  };

  // ---- node encoder: hA = x @ Wenc + benc  (WMMA)
  {
    long nx = (long)NN * 32;
    f32_to_bf16_kernel<<<cdiv(nx, B), B, 0, stream>>>(x, Abf, nx);
    pack_B_kernel<<<cdiv(32 * 64, B), B, 0, stream>>>(Wenc, Bpack, 32, 64);
    int tiles = (NN / 16) * (64 / 16);
    wmma_gemm_bf16<<<cdiv(tiles, 8), B, 0, stream>>>(Abf, Bpack, benc, hA, NN, 32, 64);
  }

  // ---- conv1: 64 -> 4x64 concat, then ELU
  run_conv(hA, 64, W1, as1, ad1, b1, 4, hB, hC);
  elu_kernel<<<cdiv((long)NN * 256, B), B, 0, stream>>>(hC, (long)NN * 256);

  // ---- conv2: 256 -> 4x64 concat, then ELU
  run_conv(hC, 256, W2, as2, ad2, b2, 4, hB, hA);
  elu_kernel<<<cdiv((long)NN * 256, B), B, 0, stream>>>(hA, (long)NN * 256);

  // ---- conv3: 256 -> 1x64 (mean over 1 head == identity)
  run_conv(hA, 256, W3, as3, ad3, b3, 1, hB, hC);

  // ---- global mean pool + decoder + softmax
  fill_kernel<<<1, B, 0, stream>>>(sums, 0.0f, GG * 64);
  fill_kernel<<<1, 64, 0, stream>>>(cnts, 0.0f, GG);
  pool_kernel<<<200, B, 0, stream>>>(hC, batch, sums, cnts, NN);
  decoder_kernel<<<1, B, 0, stream>>>(sums, cnts, gf, Wg, bg, Wd1, bd1,
                                      gamma, beta, Wd2, bd2, out);
}
